// MHA_9371618639920
// MI455X (gfx1250) — compile-verified, matrix operand order checked by reading, not run on hardware
//
#include <hip/hip_runtime.h>
#include <hip/hip_bf16.h>
#include <math.h>
#include <stdint.h>

// ---------------------------------------------------------------------------
// MHA (RoPE + causal flash attention) for MI455X / gfx1250, wave32 + WMMA.
// Matmuls run through V_WMMA_F32_16X16X32_BF16 (f32 accumulate).
// GEMM staging uses GLOBAL_LOAD_ASYNC_TO_LDS_B128 + ASYNCcnt double buffering.
// 1/sqrt(dk) is folded into Q at RoPE time; causal masking only runs on the
// diagonal k-block (uniform branch), keeping the streaming path compare-free.
// ---------------------------------------------------------------------------

typedef __attribute__((ext_vector_type(16))) __bf16 v16bf;
typedef __attribute__((ext_vector_type(8)))  __bf16 v8bf;
typedef __attribute__((ext_vector_type(8)))  float  v8f;

#define D_MODEL 1024
#define SEQ     2048
#define BATCH   2
#define HEADS   16
#define DK      64
#define BSROWS  (BATCH*SEQ)   /* 4096 */
#define LDSW    40            /* halves per LDS row (80B) -> 16B-aligned frags */

__device__ __forceinline__ v16bf combine16(v8bf lo, v8bf hi) {
  return __builtin_shufflevector(lo, hi, 0,1,2,3,4,5,6,7,8,9,10,11,12,13,14,15);
}

__device__ __forceinline__ v8f wmma_bf16(v16bf a, v16bf b, v8f c) {
  // D = A(16x32) * B(32x16) + C, f32 accumulate
  return __builtin_amdgcn_wmma_f32_16x16x32_bf16(false, a, false, b,
                                                 (short)0, c, false, false);
}

// Async global -> LDS copy (16B per lane), tracked by ASYNCcnt.
// VDST holds the LDS byte address (flat addr[31:0] maps to LDS offset).
__device__ __forceinline__ void async_copy_b128(void* lds, const void* gptr) {
  uint32_t laddr = (uint32_t)(uintptr_t)lds;
  uint64_t gaddr = (uint64_t)(uintptr_t)gptr;
  asm volatile("global_load_async_to_lds_b128 %0, %1, off"
               :: "v"(laddr), "v"(gaddr) : "memory");
}

#define WAIT_ASYNCCNT(n) asm volatile("s_wait_asynccnt %0" :: "i"(n) : "memory")

// ---------------------------------------------------------------------------
// Conversions
// ---------------------------------------------------------------------------
__global__ void cvt_f32_bf16(const float* __restrict__ in,
                             __bf16* __restrict__ out, int n) {
  int i = blockIdx.x * blockDim.x + threadIdx.x;
  if (i < n) out[i] = (__bf16)in[i];
}

// W[1024x1024] f32 -> Wt[n][k] bf16 (transposed), write-coalesced
__global__ void cvt_transpose_1k(const float* __restrict__ in,
                                 __bf16* __restrict__ out) {
  int i = blockIdx.x * blockDim.x + threadIdx.x;  // over 1M
  int kcol = i & (D_MODEL - 1);
  int nrow = i >> 10;
  out[i] = (__bf16)in[(size_t)kcol * D_MODEL + nrow];
}

// ---------------------------------------------------------------------------
// Tiled WMMA GEMM: C[M x N] = A[M x K] * Bt[N x K]^T, bf16 in, f32 acc.
// Macro tile 64x64, 4 waves (128 thr), wave tile 32x32 (2x2 WMMA accums).
// Double-buffered async LDS staging: each wave has exactly 4 async b128 ops
// per tile; "s_wait_asynccnt 4" retires the older tile while the next one is
// still in flight (async loads complete in order).
// MODE 0: QKV projection epilogue (Q,K -> [B,H,S,dk] bf16; V -> [B,H,dk,S]).
// MODE 1: plain f32 epilogue into out[M x N].
// ---------------------------------------------------------------------------
template<int MODE>
__global__ __launch_bounds__(128) void gemm64x64(
    const __bf16* __restrict__ A,
    const __bf16* __restrict__ bt0, const __bf16* __restrict__ bt1,
    const __bf16* __restrict__ bt2,
    __bf16* __restrict__ qout, __bf16* __restrict__ kout,
    __bf16* __restrict__ vtout, float* __restrict__ fout)
{
  __shared__ __bf16 lsA[2][64 * LDSW];
  __shared__ __bf16 lsB[2][64 * LDSW];

  const __bf16* Bt = (blockIdx.z == 0) ? bt0 : (blockIdx.z == 1) ? bt1 : bt2;

  const int t     = threadIdx.x;
  const int lane  = t & 31;
  const int wave  = t >> 5;
  const int wm    = wave >> 1, wn = wave & 1;
  const int lm    = lane & 15;
  const int khalf = lane >> 4;
  const int koff  = khalf * 8;
  const int rbase = blockIdx.x * 64;
  const int nbase = blockIdx.y * 64;

  const int r0 = t >> 2, c0 = t & 3;            // staging unit 0
  const int r1 = (t + 128) >> 2, c1 = t & 3;    // staging unit 1

  const v8f zero = {0.f,0.f,0.f,0.f,0.f,0.f,0.f,0.f};
  v8f acc[2][2];
  #pragma unroll
  for (int i = 0; i < 2; ++i)
    #pragma unroll
    for (int j = 0; j < 2; ++j) acc[i][j] = zero;

  // prologue: stage tile k0=0 into buffer 0
  {
    async_copy_b128(&lsA[0][r0 * LDSW + c0 * 8],
                    A + (size_t)(rbase + r0) * D_MODEL + c0 * 8);
    async_copy_b128(&lsB[0][r0 * LDSW + c0 * 8],
                    Bt + (size_t)(nbase + r0) * D_MODEL + c0 * 8);
    async_copy_b128(&lsA[0][r1 * LDSW + c1 * 8],
                    A + (size_t)(rbase + r1) * D_MODEL + c1 * 8);
    async_copy_b128(&lsB[0][r1 * LDSW + c1 * 8],
                    Bt + (size_t)(nbase + r1) * D_MODEL + c1 * 8);
  }

  for (int k0 = 0; k0 < D_MODEL; k0 += 32) {
    const int cur = (k0 >> 5) & 1;
    const int nxt = cur ^ 1;
    if (k0 + 32 < D_MODEL) {
      // issue next tile into the other buffer while this one computes
      async_copy_b128(&lsA[nxt][r0 * LDSW + c0 * 8],
                      A + (size_t)(rbase + r0) * D_MODEL + k0 + 32 + c0 * 8);
      async_copy_b128(&lsB[nxt][r0 * LDSW + c0 * 8],
                      Bt + (size_t)(nbase + r0) * D_MODEL + k0 + 32 + c0 * 8);
      async_copy_b128(&lsA[nxt][r1 * LDSW + c1 * 8],
                      A + (size_t)(rbase + r1) * D_MODEL + k0 + 32 + c1 * 8);
      async_copy_b128(&lsB[nxt][r1 * LDSW + c1 * 8],
                      Bt + (size_t)(nbase + r1) * D_MODEL + k0 + 32 + c1 * 8);
      WAIT_ASYNCCNT(4);   // current tile's 4 ops retired; next 4 in flight
    } else {
      WAIT_ASYNCCNT(0);
    }
    __syncthreads();

    v16bf af[2], bfr[2];
    #pragma unroll
    for (int i = 0; i < 2; ++i) {
      int row = wm * 32 + i * 16 + lm;
      af[i] = combine16(*(const v8bf*)&lsA[cur][row * LDSW + koff],
                        *(const v8bf*)&lsA[cur][row * LDSW + 16 + koff]);
    }
    #pragma unroll
    for (int j = 0; j < 2; ++j) {
      int col = wn * 32 + j * 16 + lm;
      bfr[j] = combine16(*(const v8bf*)&lsB[cur][col * LDSW + koff],
                         *(const v8bf*)&lsB[cur][col * LDSW + 16 + koff]);
    }
    #pragma unroll
    for (int i = 0; i < 2; ++i)
      #pragma unroll
      for (int j = 0; j < 2; ++j)
        acc[i][j] = wmma_bf16(af[i], bfr[j], acc[i][j]);
    __syncthreads();  // buffer reusable by next iteration's issue
  }

  // epilogue: C layout -> lane&15 = column, vgpr r = row (+8 for hi lanes)
  #pragma unroll
  for (int i = 0; i < 2; ++i) {
    #pragma unroll
    for (int j = 0; j < 2; ++j) {
      #pragma unroll
      for (int r = 0; r < 8; ++r) {
        int srow = rbase + wm * 32 + i * 16 + r + khalf * 8;
        int feat = nbase + wn * 32 + j * 16 + lm;
        float v = acc[i][j][r];
        if (MODE == 1) {
          fout[(size_t)srow * D_MODEL + feat] = v;
        } else {
          int b = srow >> 11, s = srow & (SEQ - 1);
          int h = feat >> 6,  d = feat & (DK - 1);
          int bh = b * HEADS + h;
          if (blockIdx.z == 0)
            qout[((size_t)bh * SEQ + s) * DK + d] = (__bf16)v;
          else if (blockIdx.z == 1)
            kout[((size_t)bh * SEQ + s) * DK + d] = (__bf16)v;
          else
            vtout[((size_t)bh * DK + d) * SEQ + s] = (__bf16)v;
        }
      }
    }
  }
}

// ---------------------------------------------------------------------------
// RoPE (interleaved pairs), f32 math on bf16 storage.
// Q additionally absorbs the attention scale 1/sqrt(dk) = 0.125, so the
// flash-attention inner loop needs no per-score multiply.
// ---------------------------------------------------------------------------
__global__ void rope_kernel(__bf16* __restrict__ q, __bf16* __restrict__ k) {
  int id = blockIdx.x * blockDim.x + threadIdx.x;  // BH*SEQ*32
  int dp   = id & 31;
  int rest = id >> 5;               // bh*SEQ + s
  int s    = rest & (SEQ - 1);
  float th = (float)s * __expf(-0.28782313662425575f * (float)dp);
  float sn, cs;
  __sincosf(th, &sn, &cs);
  size_t off = (size_t)rest * DK + 2 * dp;
  float a0 = (float)q[off], a1 = (float)q[off + 1];
  q[off]     = (__bf16)((a0 * cs - a1 * sn) * 0.125f);
  q[off + 1] = (__bf16)((a1 * cs + a0 * sn) * 0.125f);
  float b0 = (float)k[off], b1 = (float)k[off + 1];
  k[off]     = (__bf16)(b0 * cs - b1 * sn);
  k[off + 1] = (__bf16)(b1 * cs + b0 * sn);
}

// ---------------------------------------------------------------------------
// Causal flash attention: one wave per 16-row Q tile per (b,h).
// Per 32 k-positions: 4 WMMA for S = Q*K^T, online softmax (shfl_xor row
// reductions over the C layout), P -> LDS -> A-fragment, 4 WMMA for O += P*V.
// Masking is a uniform branch: only the diagonal block (kb+31 > q0) runs the
// per-element compare/cndmask chain.
// ---------------------------------------------------------------------------
__global__ __launch_bounds__(32) void attn_kernel(
    const __bf16* __restrict__ q, const __bf16* __restrict__ k,
    const __bf16* __restrict__ vt, __bf16* __restrict__ attn)
{
  __shared__ __bf16 lsP[16 * LDSW];
  const int bh    = blockIdx.y;
  const int q0    = blockIdx.x * 16;
  const int lane  = threadIdx.x;
  const int lm    = lane & 15;
  const int khalf = lane >> 4;
  const int koff  = khalf * 8;

  // Q A-fragments: dk=64 -> two 16x32 fragments, held in registers
  const __bf16* qrow = q + ((size_t)bh * SEQ + q0 + lm) * DK;
  v16bf qf[2];
  #pragma unroll
  for (int c = 0; c < 2; ++c)
    qf[c] = combine16(*(const v8bf*)(qrow + c * 32 + koff),
                      *(const v8bf*)(qrow + c * 32 + 16 + koff));

  const v8f zero = {0.f,0.f,0.f,0.f,0.f,0.f,0.f,0.f};
  v8f o[4];
  float mrow[8], lrow[8];
  #pragma unroll
  for (int j = 0; j < 4; ++j) o[j] = zero;
  #pragma unroll
  for (int r = 0; r < 8; ++r) { mrow[r] = -1e30f; lrow[r] = 0.f; }

  const int kend = q0 + 16;
  for (int kb = 0; kb < kend; kb += 32) {
    // warm L2/L0 for the next iteration's K/V tiles
    if (kb + 32 < kend) {
      __builtin_prefetch(k + ((size_t)bh * SEQ + kb + 32 + lm) * DK, 0, 1);
      __builtin_prefetch(k + ((size_t)bh * SEQ + kb + 48 + lm) * DK, 0, 1);
      __builtin_prefetch(vt + ((size_t)bh * DK + lm) * SEQ + kb + 32, 0, 1);
      __builtin_prefetch(vt + ((size_t)bh * DK + 16 + lm) * SEQ + kb + 32, 0, 1);
    }

    // scores for two 16-wide k tiles
    v8f s0 = zero, s1 = zero;
    #pragma unroll
    for (int c = 0; c < 2; ++c) {
      const __bf16* kr0 = k + ((size_t)bh * SEQ + kb + lm) * DK + c * 32;
      v16bf kf0 = combine16(*(const v8bf*)(kr0 + koff),
                            *(const v8bf*)(kr0 + 16 + koff));
      s0 = wmma_bf16(qf[c], kf0, s0);
      const __bf16* kr1 = kr0 + 16 * DK;
      v16bf kf1 = combine16(*(const v8bf*)(kr1 + koff),
                            *(const v8bf*)(kr1 + 16 + koff));
      s1 = wmma_bf16(qf[c], kf1, s1);
    }

    // causal mask: uniform branch -- only the diagonal block needs compares.
    // C layout: row = q0 + r + 8*khalf, col = kb + lm (tile0) / +16 (tile1)
    float sv0[8], sv1[8];
    if (kb + 31 <= q0) {            // fully unmasked block
      #pragma unroll
      for (int r = 0; r < 8; ++r) { sv0[r] = s0[r]; sv1[r] = s1[r]; }
    } else {                        // diagonal block
      #pragma unroll
      for (int r = 0; r < 8; ++r) {
        int rowg = q0 + r + khalf * 8;
        sv0[r] = (kb + lm      <= rowg) ? s0[r] : -1e30f;
        sv1[r] = (kb + 16 + lm <= rowg) ? s1[r] : -1e30f;
      }
    }

    // online softmax update
    float alpha[8];
    #pragma unroll
    for (int r = 0; r < 8; ++r) {
      float rm = fmaxf(sv0[r], sv1[r]);
      rm = fmaxf(rm, __shfl_xor(rm, 1, 32));
      rm = fmaxf(rm, __shfl_xor(rm, 2, 32));
      rm = fmaxf(rm, __shfl_xor(rm, 4, 32));
      rm = fmaxf(rm, __shfl_xor(rm, 8, 32));
      float mn = fmaxf(mrow[r], rm);
      alpha[r] = __expf(mrow[r] - mn);
      mrow[r] = mn;
      float e0 = __expf(sv0[r] - mn), e1 = __expf(sv1[r] - mn);
      float rs = e0 + e1;
      rs += __shfl_xor(rs, 1, 32);
      rs += __shfl_xor(rs, 2, 32);
      rs += __shfl_xor(rs, 4, 32);
      rs += __shfl_xor(rs, 8, 32);
      lrow[r] = lrow[r] * alpha[r] + rs;
      int m = r + khalf * 8;
      lsP[m * LDSW + lm]      = (__bf16)e0;
      lsP[m * LDSW + 16 + lm] = (__bf16)e1;
    }
    #pragma unroll
    for (int j = 0; j < 4; ++j)
      #pragma unroll
      for (int r = 0; r < 8; ++r) o[j][r] *= alpha[r];

    __syncthreads();  // P stores visible before A-fragment reload

    v16bf pa = combine16(*(const v8bf*)&lsP[lm * LDSW + koff],
                         *(const v8bf*)&lsP[lm * LDSW + 16 + koff]);
    #pragma unroll
    for (int j = 0; j < 4; ++j) {
      // V^T layout [bh][d][s]: column d contiguous along k positions
      const __bf16* vr = vt + ((size_t)bh * DK + j * 16 + lm) * SEQ + kb;
      v16bf vf = combine16(*(const v8bf*)(vr + koff),
                           *(const v8bf*)(vr + 16 + koff));
      o[j] = wmma_bf16(pa, vf, o[j]);
    }
    __syncthreads();  // protect lsP before next iteration overwrites
  }

  // normalize + store to [B*S, D] bf16 for the output projection
  int b = bh >> 4, h = bh & 15;
  #pragma unroll
  for (int r = 0; r < 8; ++r) {
    float inv = 1.0f / lrow[r];
    size_t srow = (size_t)b * SEQ + q0 + r + khalf * 8;
    #pragma unroll
    for (int j = 0; j < 4; ++j)
      attn[srow * D_MODEL + h * DK + j * 16 + lm] = (__bf16)(o[j][r] * inv);
  }
}

// ---------------------------------------------------------------------------
// Launch: convert -> QKV projection -> RoPE -> flash attention -> out proj.
// Workspace layout (48 MiB total):
//   xb 8MiB | wqt/wkt/wvt/wot 4x2MiB | q 8MiB | k 8MiB | vt 8MiB | attn 8MiB
// ---------------------------------------------------------------------------
extern "C" void kernel_launch(void* const* d_in, const int* in_sizes, int n_in,
                              void* d_out, int out_size, void* d_ws, size_t ws_size,
                              hipStream_t stream) {
  (void)in_sizes; (void)n_in; (void)out_size; (void)ws_size;
  const float* x  = (const float*)d_in[0];
  const float* Wq = (const float*)d_in[1];
  const float* Wk = (const float*)d_in[2];
  const float* Wv = (const float*)d_in[3];
  const float* Wo = (const float*)d_in[4];
  float* out = (float*)d_out;

  char* ws = (char*)d_ws;
  const size_t XB = (size_t)BSROWS * D_MODEL * 2;            // 8 MiB
  const size_t WB = (size_t)D_MODEL * D_MODEL * 2;           // 2 MiB
  const size_t QB = (size_t)BATCH * HEADS * SEQ * DK * 2;    // 8 MiB
  __bf16* xb    = (__bf16*)(ws);
  __bf16* wqt   = (__bf16*)(ws + XB);
  __bf16* wkt   = (__bf16*)(ws + XB + WB);
  __bf16* wvt   = (__bf16*)(ws + XB + 2 * WB);
  __bf16* wot   = (__bf16*)(ws + XB + 3 * WB);
  __bf16* qb    = (__bf16*)(ws + XB + 4 * WB);
  __bf16* kb    = (__bf16*)(ws + XB + 4 * WB + QB);
  __bf16* vtb   = (__bf16*)(ws + XB + 4 * WB + 2 * QB);
  __bf16* attnb = (__bf16*)(ws + XB + 4 * WB + 3 * QB);

  const int NX = BSROWS * D_MODEL;   // 4M
  const int NW = D_MODEL * D_MODEL;  // 1M
  cvt_f32_bf16<<<NX / 256, 256, 0, stream>>>(x, xb, NX);
  cvt_transpose_1k<<<NW / 256, 256, 0, stream>>>(Wq, wqt);
  cvt_transpose_1k<<<NW / 256, 256, 0, stream>>>(Wk, wkt);
  cvt_transpose_1k<<<NW / 256, 256, 0, stream>>>(Wv, wvt);
  cvt_transpose_1k<<<NW / 256, 256, 0, stream>>>(Wo, wot);

  dim3 gp(BSROWS / 64, D_MODEL / 64, 3);
  gemm64x64<0><<<gp, 128, 0, stream>>>(xb, wqt, wkt, wvt,
                                       qb, kb, vtb, nullptr);

  const int NR = BATCH * HEADS * SEQ * (DK / 2);  // 2M pairs
  rope_kernel<<<NR / 256, 256, 0, stream>>>(qb, kb);

  dim3 ga(SEQ / 16, BATCH * HEADS, 1);
  attn_kernel<<<ga, 32, 0, stream>>>(qb, kb, vtb, attnb);

  dim3 gf(BSROWS / 64, D_MODEL / 64, 1);
  gemm64x64<1><<<gf, 128, 0, stream>>>(attnb, wot, wot, wot,
                                       nullptr, nullptr, nullptr, out);
}
